// TemporalSAT_89309549953157
// MI455X (gfx1250) — matrix operand order repaired and checked by reading, not compiled, not run on hardware
//
#include <hip/hip_runtime.h>
#include <hip/hip_bf16.h>

typedef __attribute__((ext_vector_type(16))) _Float16 v16h;
typedef __attribute__((ext_vector_type(8)))  float    v8f;

#define TQ_B   8
#define TQ_T   2048
#define TQ_D   256     // IN_DIM == HEAD_TOTAL
#define TILE   128
#define KSTEP  32

union FragH { v16h v; uint4 q[2]; };

// ---------------------------------------------------------------------------
// CDNA5 async copy: LDS[lds_off] = MEM[saddr + voffset], 16 bytes, ASYNCcnt.
// ---------------------------------------------------------------------------
__device__ __forceinline__ void async_copy_b128(uint32_t lds_off, const void* sbase,
                                                uint32_t voffset) {
  asm volatile("global_load_async_to_lds_b128 %0, %1, %2"
               :: "v"(lds_off), "v"(voffset), "s"(sbase)
               : "memory");
}

__device__ __forceinline__ void wait_asynccnt_le4() {
  asm volatile("s_wait_asynccnt 0x4" ::: "memory");
}
__device__ __forceinline__ void wait_asynccnt_0() {
  asm volatile("s_wait_asynccnt 0x0" ::: "memory");
}

// Stage a 128x32 fp32 tile (row stride srcStride floats) into LDS as f16 [128][32].
// 256 threads: thread t handles row t/2, halves [(t&1)*16 .. +16).
__device__ __forceinline__ void stage_tile_f32_to_f16(const float* __restrict__ src,
                                                      int srcStride, _Float16* lds) {
  const int t = threadIdx.x;
  const int r = t >> 1;
  const int c = (t & 1) * 16;
  const float4* g  = (const float4*)(src + (size_t)r * srcStride + c);
  uint4*       dst = (uint4*)(lds + r * KSTEP + c);
#pragma unroll
  for (int i = 0; i < 2; ++i) {
    float4 f0 = g[2 * i + 0];
    float4 f1 = g[2 * i + 1];
    union { _Float16 h[8]; uint4 u; } pk;
    pk.h[0] = (_Float16)f0.x; pk.h[1] = (_Float16)f0.y;
    pk.h[2] = (_Float16)f0.z; pk.h[3] = (_Float16)f0.w;
    pk.h[4] = (_Float16)f1.x; pk.h[5] = (_Float16)f1.y;
    pk.h[6] = (_Float16)f1.z; pk.h[7] = (_Float16)f1.w;
    dst[i] = pk.u;
  }
}

// A fragment, 16x32 f16 (ISA layout): lanes 0-15 row m hold K 0..7 (v0-3) and
// 16..23 (v4-7); lanes 16-31 hold K 8..15 and 24..31. LDS tile is row-major [128][32].
__device__ __forceinline__ v16h load_frag_a(const _Float16* lds, int rowBase) {
  const int lane = threadIdx.x & 31;
  const int m  = lane & 15;
  const int hi = lane >> 4;
  const uint4* p = (const uint4*)(lds + (rowBase + m) * KSTEP);  // 4 chunks of 8 halves
  FragH f;
  f.q[0] = p[hi];
  f.q[1] = p[2 + hi];
  return f.v;
}

// B fragment, 32x16 f16 (ISA layout): lanes 0-15 column n hold K 0..7 and 16..23,
// lanes 16-31 hold K 8..15 and 24..31 of row n of the staged tile.
__device__ __forceinline__ v16h load_frag_b(const _Float16* lds, int rowBase) {
  const int lane = threadIdx.x & 31;
  const int n  = lane & 15;
  const int hi = lane >> 4;
  const uint4* p = (const uint4*)(lds + (rowBase + n) * KSTEP);
  FragH f;
  f.q[0] = p[hi];
  f.q[1] = p[2 + hi];
  return f.v;
}

// ---------------------------------------------------------------------------
// Kernel 1: Q = f16(x @ Wq^T), K = f16(x @ Wk^T).  blockIdx.z selects projection.
// fp32 -> f16 conversion must pass through the VALU, so staging is synchronous.
// ---------------------------------------------------------------------------
__global__ __launch_bounds__(256) void tsat_proj_kernel(const float* __restrict__ x,
                                                        const float* __restrict__ Wq,
                                                        const float* __restrict__ Wk,
                                                        _Float16* __restrict__ qh,
                                                        _Float16* __restrict__ kh) {
  __shared__ _Float16 ldsA[TILE * KSTEP];
  __shared__ _Float16 ldsB[TILE * KSTEP];

  const float* W   = (blockIdx.z == 0) ? Wq : Wk;
  _Float16*   out  = (blockIdx.z == 0) ? qh : kh;
  const int mBase  = blockIdx.y * TILE;   // rows of x (flattened B*T)
  const int nBase  = blockIdx.x * TILE;   // output feature dims
  const int w      = threadIdx.x >> 5;
  const int lane   = threadIdx.x & 31;
  const int wm     = w & 3;               // 4 waves along M (32 rows each)
  const int wn     = w >> 2;              // 2 waves along N (64 cols each)

  v8f acc[2][4];
#pragma unroll
  for (int i = 0; i < 2; ++i)
#pragma unroll
    for (int j = 0; j < 4; ++j) acc[i][j] = {};

  for (int k0 = 0; k0 < TQ_D; k0 += KSTEP) {
    __syncthreads();
    stage_tile_f32_to_f16(x + (size_t)mBase * TQ_D + k0, TQ_D, ldsA);
    stage_tile_f32_to_f16(W + (size_t)nBase * TQ_D + k0, TQ_D, ldsB);  // B[i][o] = W[o][i]
    __syncthreads();

    v16h a[2], b[4];
#pragma unroll
    for (int mt = 0; mt < 2; ++mt) a[mt] = load_frag_a(ldsA, wm * 32 + mt * 16);
#pragma unroll
    for (int nt = 0; nt < 4; ++nt) b[nt] = load_frag_b(ldsB, wn * 64 + nt * 16);
#pragma unroll
    for (int mt = 0; mt < 2; ++mt)
#pragma unroll
      for (int nt = 0; nt < 4; ++nt)
        acc[mt][nt] = __builtin_amdgcn_wmma_f32_16x16x32_f16(
            false, a[mt], false, b[nt], (short)0, acc[mt][nt], false, false);
  }

  // C/D layout: M = r + 8*hi, N = lane&15.  Store as f16.
  const int hi = lane >> 4;
  const int nn = lane & 15;
#pragma unroll
  for (int mt = 0; mt < 2; ++mt)
#pragma unroll
    for (int nt = 0; nt < 4; ++nt)
#pragma unroll
      for (int r = 0; r < 8; ++r) {
        const int row = mBase + wm * 32 + mt * 16 + r + 8 * hi;
        const int col = nBase + wn * 64 + nt * 16 + nn;
        out[(size_t)row * TQ_D + col] = (_Float16)acc[mt][nt][r];
      }
}

// ---------------------------------------------------------------------------
// Kernel 2: e[b,t,s] = (Q[b,t,:] . K[b,s,:]) / 16 + (s < t ? -1e9 : 0)
// Double-buffered LDS fed by async global->LDS copies (ASYNCcnt), so the
// stage-(k+1) copy overlaps the stage-k WMMA chain.
// ---------------------------------------------------------------------------
__global__ __launch_bounds__(256) void tsat_score_kernel(const _Float16* __restrict__ qh,
                                                         const _Float16* __restrict__ kh,
                                                         float* __restrict__ e) {
  __shared__ _Float16 ldsQ[2][TILE * KSTEP];
  __shared__ _Float16 ldsK[2][TILE * KSTEP];

  const int bb    = blockIdx.z;
  const int tBase = blockIdx.y * TILE;
  const int sBase = blockIdx.x * TILE;
  const _Float16* qsrc = qh + ((size_t)bb * TQ_T + tBase) * TQ_D;
  const _Float16* ksrc = kh + ((size_t)bb * TQ_T + sBase) * TQ_D;

  const int w    = threadIdx.x >> 5;
  const int lane = threadIdx.x & 31;
  const int wm   = w & 3;
  const int wn   = w >> 2;

  // Per-thread copy slice: row t/2, halves [(t&1)*16 .. +16)  => 2 x b128 per tile.
  const int rr = threadIdx.x >> 1;
  const int cc = (threadIdx.x & 1) * 16;

  v8f acc[2][4];
#pragma unroll
  for (int i = 0; i < 2; ++i)
#pragma unroll
    for (int j = 0; j < 4; ++j) acc[i][j] = {};

  auto issue_stage = [&](int buf, int k0) {
    const uint32_t goff = (uint32_t)(((size_t)rr * TQ_D + k0 + cc) * sizeof(_Float16));
    const uint32_t lq = (uint32_t)(uintptr_t)(&ldsQ[buf][rr * KSTEP + cc]);
    const uint32_t lk = (uint32_t)(uintptr_t)(&ldsK[buf][rr * KSTEP + cc]);
    async_copy_b128(lq,      qsrc, goff);
    async_copy_b128(lq + 16, qsrc, goff + 16);
    async_copy_b128(lk,      ksrc, goff);
    async_copy_b128(lk + 16, ksrc, goff + 16);
  };

  issue_stage(0, 0);

  for (int k = 0; k < TQ_D / KSTEP; ++k) {
    const int cur = k & 1;
    if (k < TQ_D / KSTEP - 1) {
      issue_stage(cur ^ 1, (k + 1) * KSTEP);   // prefetch next stage (other buffer)
      wait_asynccnt_le4();                     // in-order: stage-k group has landed
    } else {
      wait_asynccnt_0();
    }
    __syncthreads();                           // all waves' LDS writes visible

    v16h a[2], b[4];
#pragma unroll
    for (int mt = 0; mt < 2; ++mt) a[mt] = load_frag_a(ldsQ[cur], wm * 32 + mt * 16);
#pragma unroll
    for (int nt = 0; nt < 4; ++nt) b[nt] = load_frag_b(ldsK[cur], wn * 64 + nt * 16);
#pragma unroll
    for (int mt = 0; mt < 2; ++mt)
#pragma unroll
      for (int nt = 0; nt < 4; ++nt)
        acc[mt][nt] = __builtin_amdgcn_wmma_f32_16x16x32_f16(
            false, a[mt], false, b[nt], (short)0, acc[mt][nt], false, false);

    __syncthreads();                           // reads done before buf reused
  }

  const float scale = 0.0625f;  // 1/sqrt(256)
  const int hi = lane >> 4;
  const int nn = lane & 15;
#pragma unroll
  for (int mt = 0; mt < 2; ++mt)
#pragma unroll
    for (int nt = 0; nt < 4; ++nt)
#pragma unroll
      for (int r = 0; r < 8; ++r) {
        const int trow = tBase + wm * 32 + mt * 16 + r + 8 * hi;
        const int scol = sBase + wn * 64 + nt * 16 + nn;
        const float v = acc[mt][nt][r] * scale + ((scol < trow) ? -1e9f : 0.0f);
        e[((size_t)bb * TQ_T + trow) * TQ_T + scol] = v;
      }
}

extern "C" void kernel_launch(void* const* d_in, const int* in_sizes, int n_in,
                              void* d_out, int out_size, void* d_ws, size_t ws_size,
                              hipStream_t stream) {
  (void)in_sizes; (void)n_in; (void)out_size; (void)ws_size;
  const float* x  = (const float*)d_in[0];
  const float* Wq = (const float*)d_in[1];
  const float* Wk = (const float*)d_in[2];
  float* e = (float*)d_out;

  _Float16* qh = (_Float16*)d_ws;                         // [B*T, 256] f16
  _Float16* kh = qh + (size_t)TQ_B * TQ_T * TQ_D;         // [B*T, 256] f16

  dim3 gridProj(TQ_D / TILE, (TQ_B * TQ_T) / TILE, 2);    // (2, 128, 2)
  tsat_proj_kernel<<<gridProj, 256, 0, stream>>>(x, Wq, Wk, qh, kh);

  dim3 gridScore(TQ_T / TILE, TQ_T / TILE, TQ_B);         // (16, 16, 8)
  tsat_score_kernel<<<gridScore, 256, 0, stream>>>(qh, kh, e);
}
